// pRNN_52561809769224
// MI455X (gfx1250) — matrix operand order repaired and checked by reading, not compiled, not run on hardware
//
#include <hip/hip_runtime.h>
#include <hip/hip_bf16.h>
#include <math.h>

// Problem constants (match reference setup_inputs)
#define TLEN   8192
#define BATCH  8
#define HDIM   256
#define DIN    256
#define BM     128          // M tile per workgroup
#define BN     64           // N tile per workgroup
#define KS     64           // K slab staged in LDS
#define NCHUNK 16
#define CHUNK  (TLEN / NCHUNK)
#define NBTH   (BATCH * TLEN * HDIM)    // 16,777,216 elements

typedef __attribute__((ext_vector_type(16))) __bf16 v16bf;
typedef __attribute__((ext_vector_type(8)))  __bf16 v8bf;
typedef __attribute__((ext_vector_type(8)))  float  v8f;
typedef int v4i __attribute__((vector_size(16)));   // 4 x i32 = 128-bit payload

// ---------------- async global->LDS (guarded; falls back to ld/st) --------
#if defined(__gfx1250__) && __has_builtin(__builtin_amdgcn_global_load_async_to_lds_b128)
#define ASYNC_CP 1
#define AS_G __attribute__((address_space(1)))
#define AS_L __attribute__((address_space(3)))
#endif

__device__ __forceinline__ void async_copy_b128(const float* gsrc, float* ldst) {
#ifdef ASYNC_CP
  float* g = const_cast<float*>(gsrc);
  __builtin_amdgcn_global_load_async_to_lds_b128((AS_G v4i*)g, (AS_L v4i*)ldst, 0, 0);
#else
  *reinterpret_cast<float4*>(ldst) = *reinterpret_cast<const float4*>(gsrc);
#endif
}

__device__ __forceinline__ void async_wait0() {
#ifdef ASYNC_CP
#if __has_builtin(__builtin_amdgcn_s_wait_asynccnt)
  __builtin_amdgcn_s_wait_asynccnt(0);
#else
  asm volatile("s_wait_asynccnt 0" ::: "memory");
#endif
#endif
}

// ---------------- WMMA wrapper -------------------------------------------
__device__ __forceinline__ v8f wmma_bf16(v16bf a, v16bf b, v8f c) {
#if defined(__gfx1250__) && __has_builtin(__builtin_amdgcn_wmma_f32_16x16x32_bf16)
  return __builtin_amdgcn_wmma_f32_16x16x32_bf16(false, a, false, b, (short)0, c,
                                                 false, false);
#else
  return c; // host-pass stub
#endif
}

// A fragment: 16x32 bf16 (M x K). ISA 7.12.2: lane L holds row M = L%16;
// lanes 0-15: K = kk+[0..7] then kk+[16..23]; lanes 16-31: +8 on both chunks.
__device__ __forceinline__ v16bf frag_a(const __bf16* lds_a, int rowBase, int kk) {
  const int lane = threadIdx.x & 31;
  const int r  = rowBase + (lane & 15);
  const int kb = kk + ((lane >> 4) << 3);
  v8bf c0 = *reinterpret_cast<const v8bf*>(&lds_a[r * KS + kb]);
  v8bf c1 = *reinterpret_cast<const v8bf*>(&lds_a[r * KS + kb + 16]);
  v16bf a;
#pragma unroll
  for (int i = 0; i < 8; i++) { a[i] = c0[i]; a[i + 8] = c1[i]; }
  return a;
}

// B fragment: 32x16 bf16 (K x N), element B[k,n] = W[n,k] (row-major W,
// pre-converted to bf16 in LDS). lane L: col N = L%16; K = kk+(L/16)*8+[0..7] then +16.
// Pure ds_load, no VALU.
__device__ __forceinline__ v16bf frag_b(const __bf16* lds_wb, int ntile, int kk) {
  const int lane = threadIdx.x & 31;
  const int n  = (ntile << 4) + (lane & 15);
  const int ks = kk + ((lane >> 4) << 3);
  v8bf c0 = *reinterpret_cast<const v8bf*>(&lds_wb[n * KS + ks]);
  v8bf c1 = *reinterpret_cast<const v8bf*>(&lds_wb[n * KS + ks + 16]);
  v16bf r;
#pragma unroll
  for (int i = 0; i < 8; i++) { r[i] = c0[i]; r[i + 8] = c1[i]; }
  return r;
}

// states live in d_out at the final interleaved layout; index map folds the
// bwd time reversal so all compute kernels run in "scan time".
__device__ __forceinline__ size_t out_base(int dir, int b, int t) {
  int tt = dir ? (TLEN - 1 - t) : t;
  return ((size_t)b * TLEN + tt) * (2 * HDIM) + (size_t)dir * HDIM;
}

__device__ __forceinline__ float sigmoidf_(float x) {
  return 1.0f / (1.0f + __expf(-x));
}

// one-shot f32->bf16 slab conversion: lds_w (f32, async-filled) -> lds_wb
__device__ __forceinline__ void convert_w_slab(const float* lds_w, __bf16* lds_wb,
                                               int tid) {
#pragma unroll
  for (int i = 0; i < 4; i++) {
    int e = (tid + i * 256) << 2;
    int r = e >> 6, c = e & 63;
    float4 v = *reinterpret_cast<const float4*>(&lds_w[r * KS + c]);
    __bf16* d = &lds_wb[r * KS + c];
    d[0] = (__bf16)v.x; d[1] = (__bf16)v.y; d[2] = (__bf16)v.z; d[3] = (__bf16)v.w;
  }
}

// ============ kernel 1: z = tanh(X_dir @ W_in^T + b_in) ===================
__global__ __launch_bounds__(256) void k_gemm_z(
    const float* __restrict__ x, const float* __restrict__ Win,
    const float* __restrict__ bin, float* __restrict__ z, int dir) {
  __shared__ __align__(16) __bf16 lds_a[BM * KS];
  __shared__ __align__(16) float  lds_w[BN * KS];
  __shared__ __align__(16) __bf16 lds_wb[BN * KS];

  const int tid = threadIdx.x;
  const int lane = tid & 31;
  const int rowBase = (tid >> 5) << 4;
  const int n0 = blockIdx.y * BN;
  const int m0 = blockIdx.x * BM;

  v8f acc[4];
#pragma unroll
  for (int j = 0; j < 4; j++)
#pragma unroll
    for (int i = 0; i < 8; i++) acc[j][i] = 0.0f;

  for (int k0 = 0; k0 < DIN; k0 += KS) {
    __syncthreads();
    // stage W slab async: W[n0+r][k0+c], f32, 64x64
#pragma unroll
    for (int i = 0; i < 4; i++) {
      int e = (tid + i * 256) << 2;
      int r = e >> 6, c = e & 63;
      async_copy_b128(&Win[(size_t)(n0 + r) * DIN + k0 + c], &lds_w[r * KS + c]);
    }
    if (k0 + KS < DIN)
      __builtin_prefetch(&Win[(size_t)(n0 + (tid >> 2)) * DIN + k0 + KS + ((tid & 3) << 4)], 0, 1);
    // stage X slab through regs with f32->bf16 convert (+ time reversal)
#pragma unroll
    for (int i = 0; i < 8; i++) {
      int e = (tid + i * 256) << 2;
      int r = e >> 6, c = e & 63;
      int mg = m0 + r;
      int b = mg >> 13, t = mg & (TLEN - 1);
      int srow = b * TLEN + (dir ? (TLEN - 1 - t) : t);
      float4 v = *reinterpret_cast<const float4*>(&x[(size_t)srow * DIN + k0 + c]);
      __bf16* d = &lds_a[r * KS + c];
      d[0] = (__bf16)v.x; d[1] = (__bf16)v.y; d[2] = (__bf16)v.z; d[3] = (__bf16)v.w;
    }
    async_wait0();
    __syncthreads();
    convert_w_slab(lds_w, lds_wb, tid);   // bf16 mirror, once per slab
    __syncthreads();
#pragma unroll
    for (int kk = 0; kk < KS; kk += 32) {
      v16bf af = frag_a(lds_a, rowBase, kk);
#pragma unroll
      for (int j = 0; j < 4; j++) {
        v16bf bf = frag_b(lds_wb, j, kk);
        acc[j] = wmma_bf16(af, bf, acc[j]);
      }
    }
  }
  // epilogue (C layout: vgpr i -> M = i + 8*(lane>=16), N = lane%16)
  const int mrow0 = m0 + rowBase + ((lane >> 4) << 3);
#pragma unroll
  for (int j = 0; j < 4; j++) {
    int n = n0 + (j << 4) + (lane & 15);
    float bias = bin[n];
#pragma unroll
    for (int i = 0; i < 8; i++) {
      int m = mrow0 + i;
      z[(size_t)m * HDIM + n] = tanhf(acc[j][i] + bias);
    }
  }
}

// ====== kernel 2: u = sigm(hprev@Wrec^T + z@Uz^T + bu); emit As, bs =======
__global__ __launch_bounds__(256) void k_fused_ub(
    const float* __restrict__ z, const float* __restrict__ Wrec,
    const float* __restrict__ Uz, const float* __restrict__ bu,
    const float* __restrict__ states, float* __restrict__ Aarr,
    float* __restrict__ Barr, int dir, int use_hprev) {
  __shared__ __align__(16) __bf16 lds_a[BM * KS];
  __shared__ __align__(16) float  lds_w[BN * KS];
  __shared__ __align__(16) __bf16 lds_wb[BN * KS];

  const int tid = threadIdx.x;
  const int lane = tid & 31;
  const int rowBase = (tid >> 5) << 4;
  const int n0 = blockIdx.y * BN;
  const int m0 = blockIdx.x * BM;

  v8f acc[4];
#pragma unroll
  for (int j = 0; j < 4; j++)
#pragma unroll
    for (int i = 0; i < 8; i++) acc[j][i] = 0.0f;

  const int nph = use_hprev ? 2 : 1;
  for (int ph = 0; ph < nph; ph++) {
    const float* Wg = (ph == 0) ? Uz : Wrec;
    for (int k0 = 0; k0 < HDIM; k0 += KS) {
      __syncthreads();
#pragma unroll
      for (int i = 0; i < 4; i++) {
        int e = (tid + i * 256) << 2;
        int r = e >> 6, c = e & 63;
        async_copy_b128(&Wg[(size_t)(n0 + r) * HDIM + k0 + c], &lds_w[r * KS + c]);
      }
#pragma unroll
      for (int i = 0; i < 8; i++) {
        int e = (tid + i * 256) << 2;
        int r = e >> 6, c = e & 63;
        int mg = m0 + r;
        int b = mg >> 13, t = mg & (TLEN - 1);
        float4 v;
        if (ph == 0) {
          v = *reinterpret_cast<const float4*>(&z[(size_t)mg * HDIM + k0 + c]);
        } else if (t > 0) {
          v = *reinterpret_cast<const float4*>(&states[out_base(dir, b, t - 1) + k0 + c]);
        } else {
          v.x = 0.f; v.y = 0.f; v.z = 0.f; v.w = 0.f;
        }
        __bf16* d = &lds_a[r * KS + c];
        d[0] = (__bf16)v.x; d[1] = (__bf16)v.y; d[2] = (__bf16)v.z; d[3] = (__bf16)v.w;
      }
      async_wait0();
      __syncthreads();
      convert_w_slab(lds_w, lds_wb, tid);
      __syncthreads();
#pragma unroll
      for (int kk = 0; kk < KS; kk += 32) {
        v16bf af = frag_a(lds_a, rowBase, kk);
#pragma unroll
        for (int j = 0; j < 4; j++) {
          v16bf bf = frag_b(lds_wb, j, kk);
          acc[j] = wmma_bf16(af, bf, acc[j]);
        }
      }
    }
  }
  // epilogue: u, fs, As, bs  (A_0 = 0; h_{-1} = 0)
  const int mrow0 = m0 + rowBase + ((lane >> 4) << 3);
#pragma unroll
  for (int j = 0; j < 4; j++) {
    int n = n0 + (j << 4) + (lane & 15);
    float bun = bu[n];
    float dr  = Wrec[(size_t)n * HDIM + n];   // diag(W_rec)
#pragma unroll
    for (int i = 0; i < 8; i++) {
      int m = mrow0 + i;
      int b = m >> 13, t = m & (TLEN - 1);
      float u  = sigmoidf_(acc[j][i] + bun);
      float zt = z[(size_t)m * HDIM + n];
      float hp = 0.0f;
      if (use_hprev && t > 0) hp = states[out_base(dir, b, t - 1) + n];
      float fs = u * hp + (1.0f - u) * zt;
      float As = (t == 0) ? 0.0f : fmaf((hp - zt) * u * (1.0f - u), dr, u);
      float bs = fs - As * hp;
      size_t o = (size_t)m * HDIM + n;
      Aarr[o] = As;
      Barr[o] = bs;
    }
  }
}

// ========== scan phase 1: per-chunk combine (carry = (prodA, b)) ==========
__global__ __launch_bounds__(256) void k_scan_reduce(
    const float* __restrict__ A, const float* __restrict__ Bv,
    float* __restrict__ cA, float* __restrict__ cB) {
  int id = blockIdx.x * 256 + threadIdx.x;       // BATCH*NCHUNK*HDIM threads
  int h = id & (HDIM - 1);
  int rest = id >> 8;
  int chunk = rest & (NCHUNK - 1);
  int b = rest >> 4;
  size_t base = ((size_t)b * TLEN + (size_t)chunk * CHUNK) * HDIM + h;
  float ca = 1.0f, cb = 0.0f;
  for (int t = 0; t < CHUNK; t++) {
    float at = A[base + (size_t)t * HDIM];
    float bt = Bv[base + (size_t)t * HDIM];
    ca = at * ca;
    cb = fmaf(at, cb, bt);
  }
  size_t o = ((size_t)b * NCHUNK + chunk) * HDIM + h;
  cA[o] = ca;
  cB[o] = cb;
}

// ========== scan phase 2: exclusive scan of chunk carries =================
__global__ __launch_bounds__(256) void k_scan_carry(
    const float* __restrict__ cA, const float* __restrict__ cB,
    float* __restrict__ pref) {
  int id = blockIdx.x * 256 + threadIdx.x;       // BATCH*HDIM threads
  int h = id & (HDIM - 1);
  int b = id >> 8;
  float pa = 1.0f, pb = 0.0f;
  for (int c = 0; c < NCHUNK; c++) {
    size_t o = ((size_t)b * NCHUNK + c) * HDIM + h;
    pref[o] = pb;                                 // state entering this chunk
    float ca = cA[o], cb = cB[o];
    pb = fmaf(ca, pb, cb);
    pa = ca * pa;
  }
}

// ========== scan phase 3: apply prefixes, emit states into d_out ==========
__global__ __launch_bounds__(256) void k_scan_apply(
    const float* __restrict__ A, const float* __restrict__ Bv,
    const float* __restrict__ pref, float* __restrict__ out, int dir) {
  int id = blockIdx.x * 256 + threadIdx.x;
  int h = id & (HDIM - 1);
  int rest = id >> 8;
  int chunk = rest & (NCHUNK - 1);
  int b = rest >> 4;
  size_t base = ((size_t)b * TLEN + (size_t)chunk * CHUNK) * HDIM + h;
  float hcur = pref[((size_t)b * NCHUNK + chunk) * HDIM + h];
  for (int t = 0; t < CHUNK; t++) {
    float at = A[base + (size_t)t * HDIM];
    float bt = Bv[base + (size_t)t * HDIM];
    hcur = fmaf(at, hcur, bt);
    out[out_base(dir, b, chunk * CHUNK + t) + h] = hcur;
  }
}

// ========== h_n gather =====================================================
__global__ __launch_bounds__(256) void k_hn(const float* __restrict__ out,
                                            float* __restrict__ hn) {
  int id = blockIdx.x * 256 + threadIdx.x;       // 2*BATCH*HDIM threads
  int h = id & (HDIM - 1);
  int b = (id >> 8) & (BATCH - 1);
  int part = id >> 11;
  float v = (part == 0)
                ? out[((size_t)b * TLEN + (TLEN - 1)) * (2 * HDIM) + h]        // fwd[:, -1]
                : out[((size_t)b * TLEN + 0) * (2 * HDIM) + HDIM + h];         // bwd[:, 0]
  hn[(size_t)part * BATCH * HDIM + (size_t)b * HDIM + h] = v;
}

// ===========================================================================
extern "C" void kernel_launch(void* const* d_in, const int* in_sizes, int n_in,
                              void* d_out, int out_size, void* d_ws, size_t ws_size,
                              hipStream_t stream) {
  (void)in_sizes; (void)n_in; (void)out_size; (void)ws_size;

  const float* x = (const float*)d_in[0];
  const float* W_in[2]  = {(const float*)d_in[1], (const float*)d_in[6]};
  const float* b_in[2]  = {(const float*)d_in[2], (const float*)d_in[7]};
  const float* W_rec[2] = {(const float*)d_in[3], (const float*)d_in[8]};
  const float* U_z[2]   = {(const float*)d_in[4], (const float*)d_in[9]};
  const float* b_u[2]   = {(const float*)d_in[5], (const float*)d_in[10]};

  float* out = (float*)d_out;
  float* hn  = out + (size_t)BATCH * TLEN * 2 * HDIM;

  // workspace: z | A | b | chunk carries (needs ~202 MB)
  float* z    = (float*)d_ws;
  float* Aarr = z + (size_t)NBTH;
  float* Barr = Aarr + (size_t)NBTH;
  float* cA   = Barr + (size_t)NBTH;
  float* cB   = cA + (size_t)BATCH * NCHUNK * HDIM;
  float* pref = cB + (size_t)BATCH * NCHUNK * HDIM;

  dim3 blk(256);
  dim3 gmm(BATCH * TLEN / BM, HDIM / BN);
  int gscan = BATCH * NCHUNK * HDIM / 256;

  for (int dir = 0; dir < 2; dir++) {
    k_gemm_z<<<gmm, blk, 0, stream>>>(x, W_in[dir], b_in[dir], z, dir);
    for (int it = 0; it < 2; it++) {
      k_fused_ub<<<gmm, blk, 0, stream>>>(z, W_rec[dir], U_z[dir], b_u[dir],
                                          out, Aarr, Barr, dir, it);
      k_scan_reduce<<<gscan, blk, 0, stream>>>(Aarr, Barr, cA, cB);
      k_scan_carry<<<BATCH * HDIM / 256, blk, 0, stream>>>(cA, cB, pref);
      k_scan_apply<<<gscan, blk, 0, stream>>>(Aarr, Barr, pref, out, dir);
    }
  }
  k_hn<<<(2 * BATCH * HDIM) / 256, blk, 0, stream>>>(out, hn);
}
